// GroupedQueryAttention_88562225643870
// MI455X (gfx1250) — compile-verified, matrix operand order checked by reading, not run on hardware
//
#include <hip/hip_runtime.h>

// ---------------- problem constants ----------------
constexpr int BB  = 2;     // batch
constexpr int T   = 2048;  // sequence
constexpr int C   = 2048;  // embed dim
constexpr int H   = 32;    // query heads
constexpr int G   = 8;     // kv groups
constexpr int HD  = 64;    // head dim
constexpr int MTOT = BB * T;
constexpr float SCALE = 0.125f;              // 64^-0.5
constexpr float LOG2E = 1.44269504088896f;

constexpr int CH  = 64;    // attention key-chunk
constexpr int PAD = 72;    // LDS row pitch in u16 (144B: 16B-aligned, bank-spread)

typedef unsigned short u16;
typedef __attribute__((ext_vector_type(16))) __bf16 v16bf;
typedef __attribute__((ext_vector_type(8)))  float  v8f;

// ---------------- scalar helpers ----------------
__device__ __forceinline__ unsigned short f2bfu(float f) {
    unsigned u = __builtin_bit_cast(unsigned, f);
    return (unsigned short)((u + 0x7FFFu + ((u >> 16) & 1u)) >> 16); // RNE
}
__device__ __forceinline__ v8f vzero8() {
    v8f v;
#pragma unroll
    for (int i = 0; i < 8; ++i) v[i] = 0.0f;
    return v;
}

// xor-butterfly reductions across the 16 lanes sharing a row-set.
#define SWZ_XOR(v, m)                                                        \
    __builtin_bit_cast(float, __builtin_amdgcn_ds_swizzle(                   \
        __builtin_bit_cast(int, v), ((m) << 10) | 0x1f))
__device__ __forceinline__ float xor_max16(float v) {
    v = fmaxf(v, SWZ_XOR(v, 1));
    v = fmaxf(v, SWZ_XOR(v, 2));
    v = fmaxf(v, SWZ_XOR(v, 4));
    v = fmaxf(v, SWZ_XOR(v, 8));
    return v;
}
__device__ __forceinline__ float xor_add16(float v) {
    v += SWZ_XOR(v, 1);
    v += SWZ_XOR(v, 2);
    v += SWZ_XOR(v, 4);
    v += SWZ_XOR(v, 8);
    return v;
}

// ---------------- fragment helpers ----------------
// A/B fragment layout (16-bit, 16x32): each lane owns two contiguous
// 8-element chunks at +0 and +16 elements from its per-lane base.
struct U16x16 { uint4 a, b; };
__device__ __forceinline__ v16bf frag_bf16(const u16* p) {
    U16x16 t;
    t.a = *(const uint4*)(p);
    t.b = *(const uint4*)(p + 16);
    return __builtin_bit_cast(v16bf, t);
}
__device__ __forceinline__ v8f wmma_bf16(v16bf a, v16bf b, v8f c) {
    return __builtin_amdgcn_wmma_f32_16x16x32_bf16(
        false, a, false, b, (short)0, c, false, false);
}

// ---------------- async global->LDS copy (CDNA5 path) ----------------
__device__ __forceinline__ void cp16_async(unsigned ldsOff, const u16* g) {
    asm volatile("global_load_async_to_lds_b128 %0, %1, off"
                 :: "v"(ldsOff), "v"(g) : "memory");
}
__device__ __forceinline__ void wait_async0() {
#if __has_builtin(__builtin_amdgcn_s_wait_asynccnt)
    __builtin_amdgcn_s_wait_asynccnt(0);
#else
    asm volatile("s_wait_asynccnt 0" ::: "memory");
#endif
}
__device__ __forceinline__ void wait_ds0() {
#if __has_builtin(__builtin_amdgcn_s_wait_dscnt)
    __builtin_amdgcn_s_wait_dscnt(0);
#else
    asm volatile("s_wait_dscnt 0" ::: "memory");
#endif
}

// ---------------- fp32 -> bf16 pre-conversion (bandwidth-bound, one pass) ---
__global__ void __launch_bounds__(256)
cvt_bf16(const float* __restrict__ in, u16* __restrict__ out) {
    const int i = blockIdx.x * 256 + threadIdx.x;   // one 8-element unit each
    const float4* p = (const float4*)in + (size_t)i * 2;
    const float4 x0 = p[0], x1 = p[1];
    uint4 o;
    o.x = f2bfu(x0.x) | ((unsigned)f2bfu(x0.y) << 16);
    o.y = f2bfu(x0.z) | ((unsigned)f2bfu(x0.w) << 16);
    o.z = f2bfu(x1.x) | ((unsigned)f2bfu(x1.y) << 16);
    o.w = f2bfu(x1.z) | ((unsigned)f2bfu(x1.w) << 16);
    ((uint4*)out)[i] = o;
}

// ---------------- Linear: out = alpha*(x @ W^T + b), all-bf16 operands ------
// Wave tile 32x64 (2 A frags, 8 accums): 12 b128 loads per 8 WMMAs.
// MODE 0: bf16 out, layout [B][H][T][64]          (Q, alpha=SCALE)
// MODE 1: bf16 out, layout [B][G][T][64]          (K)
// MODE 2: bf16 out, layout [B][G][64][T] (transp) (V)
// MODE 3: fp32 out, layout [M][N]                 (output proj)
template <int MODE>
__global__ void __launch_bounds__(128)
linear_wmma(const u16* __restrict__ X, const u16* __restrict__ W,
            const float* __restrict__ bias, void* __restrict__ Ov, float alpha) {
    constexpr int K = C;

    const int lane = threadIdx.x & 31;
    const int wv   = threadIdx.x >> 5;
    const int col  = lane & 15;
    const int hi   = lane >> 4;

    const int m0 = blockIdx.x * 128 + wv * 32;  // wave: 32 rows
    const int n0 = blockIdx.y * 64;             // wave: 64 cols

    auto LA = [&](int k0, int mi) -> v16bf {
        return frag_bf16(X + (size_t)(m0 + mi * 16 + col) * K + k0 + hi * 8);
    };
    auto LB = [&](int k0, int s) -> v16bf {
        return frag_bf16(W + (size_t)(n0 + s * 16 + col) * K + k0 + hi * 8);
    };

    v8f acc[2][4];
#pragma unroll
    for (int mi = 0; mi < 2; ++mi)
#pragma unroll
        for (int s = 0; s < 4; ++s) acc[mi][s] = vzero8();

    // software-pipelined K loop
    v16bf a[2] = {LA(0, 0), LA(0, 1)}, bfr[4];
#pragma unroll
    for (int s = 0; s < 4; ++s) bfr[s] = LB(0, s);

    for (int k0 = 0; k0 < K; k0 += 32) {
        v16bf an[2], bn[4];
        const bool more = (k0 + 32) < K;
        if (more) {
            an[0] = LA(k0 + 32, 0);
            an[1] = LA(k0 + 32, 1);
#pragma unroll
            for (int s = 0; s < 4; ++s) bn[s] = LB(k0 + 32, s);
        }
#pragma unroll
        for (int s = 0; s < 4; ++s) {
            acc[0][s] = wmma_bf16(a[0], bfr[s], acc[0][s]);
            acc[1][s] = wmma_bf16(a[1], bfr[s], acc[1][s]);
        }
        if (more) {
            a[0] = an[0]; a[1] = an[1];
#pragma unroll
            for (int s = 0; s < 4; ++s) bfr[s] = bn[s];
        }
    }

#pragma unroll
    for (int mi = 0; mi < 2; ++mi) {
        const int mb   = m0 + mi * 16;
        const int bidx = mb / T;
        const int tb   = mb % T;
#pragma unroll
        for (int s = 0; s < 4; ++s) {
            const int n  = n0 + s * 16 + col;
            const float bsv = bias[n];
            if constexpr (MODE == 2) {
                // transposed bf16 store: rows r -> consecutive t, one b128
                const int gidx = n >> 6, d = n & 63;
                unsigned pw[4];
#pragma unroll
                for (int r = 0; r < 4; ++r) {
                    unsigned lo_ = f2bfu((acc[mi][s][2 * r]     + bsv) * alpha);
                    unsigned hi_ = f2bfu((acc[mi][s][2 * r + 1] + bsv) * alpha);
                    pw[r] = lo_ | (hi_ << 16);
                }
                uint4 pk; pk.x = pw[0]; pk.y = pw[1]; pk.z = pw[2]; pk.w = pw[3];
                size_t off = ((size_t)(bidx * G + gidx) * HD + d) * T + tb + hi * 8;
                *(uint4*)((u16*)Ov + off) = pk;
            } else {
#pragma unroll
                for (int r = 0; r < 8; ++r) {
                    const float v = (acc[mi][s][r] + bsv) * alpha;
                    const int t = tb + r + 8 * hi;
                    if constexpr (MODE == 0) {
                        ((u16*)Ov)[((size_t)(bidx * H + (n >> 6)) * T + t) * HD + (n & 63)] = f2bfu(v);
                    } else if constexpr (MODE == 1) {
                        ((u16*)Ov)[((size_t)(bidx * G + (n >> 6)) * T + t) * HD + (n & 63)] = f2bfu(v);
                    } else { // MODE 3
                        ((float*)Ov)[(size_t)(mb + r + 8 * hi) * C + n] = v;
                    }
                }
            }
        }
    }
}

// ---------------- flash attention -------------------------------------------
// grid: (T/64, H, B), block: 128 (4 waves); wave w owns a 16-query tile.
// K/V 64-key chunks staged global->LDS with async b128 copies, double-buffered.
__global__ void __launch_bounds__(128)
attn_wmma(const u16* __restrict__ Qb, const u16* __restrict__ Kb,
          const u16* __restrict__ Vt, u16* __restrict__ Ab) {
    __shared__ __align__(16) u16 smem[4 * 64 * PAD + 4 * 16 * PAD];
    const unsigned smemBase = (unsigned)(unsigned long long)(const void*)smem;
    constexpr int VOFF = 2 * 64 * PAD;
    constexpr int POFF = 4 * 64 * PAD;

    const int lane = threadIdx.x & 31;
    const int wv   = threadIdx.x >> 5;
    const int col  = lane & 15;
    const int hi   = lane >> 4;

    const int q0 = blockIdx.x * 64 + wv * 16;
    const int h  = blockIdx.y;
    const int b  = blockIdx.z;
    const int g  = h >> 2; // GQA repeat_interleave(4)

    const u16* qrow = Qb + ((size_t)(b * H + h) * T + (q0 + col)) * HD;
    const v16bf aq0 = frag_bf16(qrow + 0  + hi * 8);
    const v16bf aq1 = frag_bf16(qrow + 32 + hi * 8);

    const u16* kg = Kb + (size_t)(b * G + g) * T * HD;  // [T][64]
    const u16* vg = Vt + (size_t)(b * G + g) * HD * T;  // [64][T]

    auto stage = [&](int sbuf, int kcn) {
#pragma unroll
        for (int u = 0; u < 4; ++u) {
            const int idx  = threadIdx.x + u * 128;
            const int row  = idx >> 3;
            const int in16 = idx & 7;
            cp16_async(smemBase + (unsigned)((sbuf * 64 + row) * PAD * 2 + in16 * 16),
                       kg + (size_t)(kcn + row) * HD + in16 * 8);
            cp16_async(smemBase + (unsigned)(VOFF * 2 + (sbuf * 64 + row) * PAD * 2 + in16 * 16),
                       vg + (size_t)row * T + kcn + in16 * 8);
        }
    };

    v8f o[4];
#pragma unroll
    for (int dt = 0; dt < 4; ++dt) o[dt] = vzero8();
    float mrow[8], srow[8];
#pragma unroll
    for (int r = 0; r < 8; ++r) { mrow[r] = -__builtin_inff(); srow[r] = 0.0f; }

    u16* Pw = smem + POFF + wv * 16 * PAD;

    stage(0, 0);
    for (int kc = 0; kc < T; kc += CH) {
        const int buf = (kc >> 6) & 1;
        wait_async0();
        wait_ds0();
        __syncthreads();
        if (kc + CH < T) stage(buf ^ 1, kc + CH);

        const u16* Ksh = smem + buf * 64 * PAD;
        const u16* Vsh = smem + VOFF + buf * 64 * PAD;

        // ---- S = Q*K^T : 16 x 64 scores ----
        v8f s[4];
#pragma unroll
        for (int nt = 0; nt < 4; ++nt) {
            const u16* kr = Ksh + (size_t)(nt * 16 + col) * PAD;
            v8f t = vzero8();
            t = wmma_bf16(aq0, frag_bf16(kr + hi * 8), t);
            t = wmma_bf16(aq1, frag_bf16(kr + 32 + hi * 8), t);
            s[nt] = t;
        }

        // ---- online softmax over the 64-key chunk ----
#pragma unroll
        for (int r = 0; r < 8; ++r) {
            float mx = fmaxf(fmaxf(s[0][r], s[1][r]), fmaxf(s[2][r], s[3][r]));
            mx = xor_max16(mx);
            const float mnew = fmaxf(mrow[r], mx);
            const float corr = exp2f((mrow[r] - mnew) * LOG2E);
            mrow[r] = mnew;
            float p[4], ps = 0.0f;
#pragma unroll
            for (int nt = 0; nt < 4; ++nt) {
                p[nt] = exp2f((s[nt][r] - mnew) * LOG2E);
                ps += p[nt];
            }
            ps = xor_add16(ps);
            srow[r] = srow[r] * corr + ps;
#pragma unroll
            for (int dt = 0; dt < 4; ++dt) o[dt][r] *= corr;
#pragma unroll
            for (int nt = 0; nt < 4; ++nt)
                Pw[(size_t)(r + 8 * hi) * PAD + nt * 16 + col] = f2bfu(p[nt]);
        }

        // ---- O += P * V ----
        const v16bf ap0 = frag_bf16(Pw + (size_t)col * PAD + hi * 8);
        const v16bf ap1 = frag_bf16(Pw + (size_t)col * PAD + 32 + hi * 8);
#pragma unroll
        for (int dt = 0; dt < 4; ++dt) {
            const u16* vr = Vsh + (size_t)(dt * 16 + col) * PAD;
            o[dt] = wmma_bf16(ap0, frag_bf16(vr + hi * 8), o[dt]);
            o[dt] = wmma_bf16(ap1, frag_bf16(vr + 32 + hi * 8), o[dt]);
        }
    }

    float rinv[8];
#pragma unroll
    for (int r = 0; r < 8; ++r) rinv[r] = 1.0f / srow[r];
#pragma unroll
    for (int dt = 0; dt < 4; ++dt) {
#pragma unroll
        for (int r = 0; r < 8; ++r) {
            const int q = q0 + r + 8 * hi;
            Ab[(size_t)(b * T + q) * C + h * HD + dt * 16 + col] =
                f2bfu(o[dt][r] * rinv[r]);
        }
    }
}

// ---------------- host-side launch ----------------
extern "C" void kernel_launch(void* const* d_in, const int* in_sizes, int n_in,
                              void* d_out, int out_size, void* d_ws, size_t ws_size,
                              hipStream_t stream) {
    const float* query = (const float*)d_in[0];
    const float* key   = (const float*)d_in[1];
    const float* value = (const float*)d_in[2];
    const float* q_w   = (const float*)d_in[3];
    const float* q_b   = (const float*)d_in[4];
    const float* k_w   = (const float*)d_in[5];
    const float* k_b   = (const float*)d_in[6];
    const float* v_w   = (const float*)d_in[7];
    const float* v_b   = (const float*)d_in[8];
    const float* o_w   = (const float*)d_in[9];
    const float* o_b   = (const float*)d_in[10];

    // workspace (bf16): Qb[B][H][T][64]  Kb[B][G][T][64]  Vt[B][G][64][T]
    //                   Ab[B][T][C]      Xb (activation staging, reused)
    //                   Wb (weight staging, reused)
    const size_t QE = (size_t)BB * H * T * HD;  // 8.39M
    const size_t KE = (size_t)BB * G * T * HD;  // 2.10M
    const size_t XE = (size_t)MTOT * C;         // 8.39M
    const size_t WE = (size_t)C * C;            // 4.19M
    char* ws = (char*)d_ws;
    u16* Qb = (u16*)ws;
    u16* Kb = (u16*)(ws + QE * 2);
    u16* Vt = (u16*)(ws + (QE + KE) * 2);
    u16* Ab = (u16*)(ws + (QE + 2 * KE) * 2);
    u16* Xb = (u16*)(ws + (QE + 2 * KE + QE) * 2);
    u16* Wb = (u16*)(ws + (QE + 2 * KE + QE + XE) * 2);

    const dim3 blk(128);
    const int cvtX = (int)(XE / 8 / 256);                 // activation converts
    const int cvtWqo = (int)(WE / 8 / 256);               // 2048x2048 weights
    const int cvtWkv = (int)((size_t)G * HD * C / 8 / 256);

    // Q = SCALE * (query @ q_w^T + q_b)
    cvt_bf16<<<cvtX, 256, 0, stream>>>(query, Xb);
    cvt_bf16<<<cvtWqo, 256, 0, stream>>>(q_w, Wb);
    linear_wmma<0><<<dim3(MTOT / 128, C / 64), blk, 0, stream>>>(Xb, Wb, q_b, Qb, SCALE);
    // K
    cvt_bf16<<<cvtX, 256, 0, stream>>>(key, Xb);
    cvt_bf16<<<cvtWkv, 256, 0, stream>>>(k_w, Wb);
    linear_wmma<1><<<dim3(MTOT / 128, (G * HD) / 64), blk, 0, stream>>>(Xb, Wb, k_b, Kb, 1.0f);
    // V (transposed output)
    cvt_bf16<<<cvtX, 256, 0, stream>>>(value, Xb);
    cvt_bf16<<<cvtWkv, 256, 0, stream>>>(v_w, Wb);
    linear_wmma<2><<<dim3(MTOT / 128, (G * HD) / 64), blk, 0, stream>>>(Xb, Wb, v_b, Vt, 1.0f);
    // attention
    attn_wmma<<<dim3(T / 64, H, BB), blk, 0, stream>>>(Qb, Kb, Vt, Ab);
    // out = Ab @ o_w^T + o_b  (fp32 out)
    cvt_bf16<<<cvtWqo, 256, 0, stream>>>(o_w, Wb);
    linear_wmma<3><<<dim3(MTOT / 128, C / 64), blk, 0, stream>>>(Ab, Wb, o_b, d_out, 1.0f);
}
